// BidirLSTMLayer_695784702523
// MI455X (gfx1250) — matrix operand order repaired and checked by reading, not compiled, not run on hardware
//
#include <hip/hip_runtime.h>
#include <hip/hip_bf16.h>

// Problem constants (from reference): T=512, B=64, I=256, H=512
constexpr int kT = 512;
constexpr int kB = 64;
constexpr int kI = 256;
constexpr int kH = 512;
constexpr int kG = 4 * kH;        // 2048 gate columns
constexpr int kK = kI + kH;       // 768 fused K for [x|h] @ [W_ih;W_hh]
constexpr int NBLK = 32;          // persistent blocks (must all be resident)
constexpr int TPB  = 256;         // 8 waves per block

// dynamic LDS: [64 rows x 768] fused [x|h] + [64 rows x 512] c, f16
constexpr int kXHrow = kK;                    // 768
constexpr int kCrow  = kH;                    // 512
constexpr size_t kLdsHalves = (size_t)kB * kXHrow + (size_t)kB * kCrow;
constexpr size_t kLdsBytes  = kLdsHalves * 2; // 160 KB (CDNA5 WGP has 320 KB)

typedef __attribute__((ext_vector_type(16))) _Float16 v16h;
typedef __attribute__((ext_vector_type(8)))  _Float16 v8h;
typedef __attribute__((ext_vector_type(8)))  float    v8f;

// ---------------- prep kernels ----------------

__global__ void k_cvt_x(const float* __restrict__ x, _Float16* __restrict__ y, int n) {
    int i = blockIdx.x * blockDim.x + threadIdx.x;
    if (i < n) y[i] = (_Float16)x[i];
}

// Build W_cat^T per direction: wt[dir][n][k] = (k<I ? W_ih[k][n] : W_hh[k-I][n])
__global__ void k_pack_wcat(const float* __restrict__ Wih_f, const float* __restrict__ Whh_f,
                            const float* __restrict__ Wih_b, const float* __restrict__ Whh_b,
                            _Float16* __restrict__ wt) {
    long i = (long)blockIdx.x * blockDim.x + threadIdx.x;
    const long per = (long)kG * kK;
    if (i >= 2 * per) return;
    int d = (int)(i / per);
    long r = i % per;
    int n = (int)(r / kK);
    int k = (int)(r % kK);
    const float* Wih = d ? Wih_b : Wih_f;
    const float* Whh = d ? Whh_b : Whh_f;
    float v = (k < kI) ? Wih[(long)k * kG + n] : Whh[(long)(k - kI) * kG + n];
    wt[i] = (_Float16)v;
}

// W_d^T per direction: wt[dir][n][k] = W_d[k][n]
__global__ void k_pack_wd(const float* __restrict__ Wd_f, const float* __restrict__ Wd_b,
                          _Float16* __restrict__ wt) {
    long i = (long)blockIdx.x * blockDim.x + threadIdx.x;
    const long per = (long)kH * kH;
    if (i >= 2 * per) return;
    int d = (int)(i / per);
    long r = i % per;
    int n = (int)(r / kH);
    int k = (int)(r % kH);
    const float* W = d ? Wd_b : Wd_f;
    wt[i] = (_Float16)W[(long)k * kH + n];
}

__global__ void k_zero32(unsigned* __restrict__ p, int nwords) {
    int i = blockIdx.x * blockDim.x + threadIdx.x;
    if (i < nwords) p[i] = 0u;
}

// ---------------- WMMA helpers ----------------

// A fragment (16-bit 16x32, MxK): lane supplies row (l&15); K elements
// {k0..k0+7, k0+16..k0+23}, k0 = (l>=16 ? 8 : 0) relative to chunk base.
__device__ inline v16h load_a(const _Float16* p) {
    union { v16h v; v8h h[2]; } u;
    u.h[0] = *(const v8h*)(p);
    u.h[1] = *(const v8h*)(p + 16);
    return u.v;
}

// B fragment (16-bit 32x16, KxN): lane supplies col (l&15); K elements
// {k0..k0+7, k0+16..k0+23}, k0 = (l>=16 ? 8 : 0) relative to chunk base, from W^T row.
__device__ inline v16h load_b(const _Float16* p) {
    union { v16h v; v8h h[2]; } u;
    u.h[0] = *(const v8h*)(p);
    u.h[1] = *(const v8h*)(p + 16);
    return u.v;
}

__device__ inline v8f wmma32(v16h a, v16h b, v8f c) {
    return __builtin_amdgcn_wmma_f32_16x16x32_f16(false, a, false, b, (short)0, c, false, false);
}

__device__ inline float sigmoidf_(float x) { return 1.0f / (1.0f + expf(-x)); }

// ---------------- persistent bidirectional TLSTM scan ----------------
// Block bb: dir = bb>>4, hp = bb&15  -> h column blocks {2hp, 2hp+1}
// Wave w (0..7): rb = w>>1 (batch rows rb*16..+15), hb = 2hp + (w&1)
// => the 4 rb-waves sharing a weight column sit on the SAME WGP (near-cache reuse),
//    A operands for all 64 batch rows staged in LDS once per block.
// Split grid barrier: arrive right after state stores, x-stage overlaps the wait.

__global__ __launch_bounds__(TPB, 2) void k_lstm(
    const float* __restrict__ tvec,
    const float* __restrict__ b_f,  const float* __restrict__ bd_f,
    const float* __restrict__ b_b,  const float* __restrict__ bd_b,
    const _Float16* __restrict__ x16,     // [T,B,I] f16
    const _Float16* __restrict__ wcat,    // [2][2048][768] f16 (W^T)
    const _Float16* __restrict__ wd,      // [2][512][512]  f16 (W^T)
    _Float16* __restrict__ h16,           // [2][2][B*H] f16 ping-pong
    _Float16* __restrict__ c16,           // [2][2][B*H] f16 ping-pong
    float*    __restrict__ c32,           // [2][2][B*H] f32 ping-pong
    float*    __restrict__ out,           // [T,B,2H] f32
    unsigned* __restrict__ cnt)
{
    extern __shared__ _Float16 smem[];
    _Float16* sXH = smem;                         // [64][768]
    _Float16* sC  = smem + (size_t)kB * kXHrow;   // [64][512]

    const int tid  = threadIdx.x;
    const int lane = tid & 31;
    const int w    = tid >> 5;          // wave in block, 0..7
    const int bb   = blockIdx.x;        // 0..31
    const int dir  = bb >> 4;
    const int hp   = bb & 15;
    const int rb   = w >> 1;            // 0..3
    const int hb   = hp * 2 + (w & 1);  // 0..31

    const int lm  = lane & 15;
    const int up  = lane >> 4;
    const int kbA = up * 8;
    const int kbB = up * 8;

    const _Float16* Wc  = wcat + (size_t)dir * kG * kK;
    const _Float16* Wdp = wd   + (size_t)dir * kH * kH;
    const float* bias  = dir ? b_b  : b_f;
    const float* biasd = dir ? bd_b : bd_f;

    _Float16* H16 = h16 + (size_t)dir * (2 * kB * kH);
    _Float16* C16 = c16 + (size_t)dir * (2 * kB * kH);
    float*    C32 = c32 + (size_t)dir * (2 * kB * kH);

    const int arow = rb * 16 + lm;      // batch row this lane supplies for A
    const int nlm  = hb * 16 + lm;      // output h-column this lane owns

    const _Float16* bip0 = Wc + (size_t)(0 * kH + nlm) * kK + kbB;
    const _Float16* bfp0 = Wc + (size_t)(1 * kH + nlm) * kK + kbB;
    const _Float16* bgp0 = Wc + (size_t)(2 * kH + nlm) * kK + kbB;
    const _Float16* bop0 = Wc + (size_t)(3 * kH + nlm) * kK + kbB;
    const _Float16* bdp0 = Wdp + (size_t)nlm * kH + kbB;

    const float bv_i = bias[0 * kH + nlm];
    const float bv_f = bias[1 * kH + nlm];
    const float bv_g = bias[2 * kH + nlm];
    const float bv_o = bias[3 * kH + nlm];
    const float bv_d = biasd[nlm];

    const int aoff = arow * kXHrow + kbA;   // LDS offset for gate A fragments
    const int coff = arow * kCrow  + kbA;   // LDS offset for c_s A fragments

    // x_t stage into LDS (no cross-block dependency)
    auto copy_x = [&](int tf) {
        #pragma unroll
        for (int it = 0; it < 8; ++it) {
            int idx = it * TPB + tid;             // 0..2047: 64 rows x 32 v8h
            int row = idx >> 5, c8 = idx & 31;
            *(v8h*)(sXH + row * kXHrow + c8 * 8) =
                *(const v8h*)(x16 + ((size_t)tf * kB + row) * kI + c8 * 8);
        }
    };

    copy_x(dir ? (kT - 1) : 0);

    for (int s = 0; s < kT; ++s) {
        const int tf = dir ? (kT - 1 - s) : s;
        const int p  = s & 1;

        // ---- wait: all blocks finished step s-1 (h/c ping-pong buffers ready)
        if (tid == 0) {
            const unsigned target = (unsigned)NBLK * (unsigned)s;
            while (__hip_atomic_load(cnt, __ATOMIC_RELAXED, __HIP_MEMORY_SCOPE_AGENT) < target)
                __builtin_amdgcn_s_sleep(2);
            __threadfence();             // acquire
        }
        __syncthreads();

        // ---- stage h and c into LDS (all 64 batch rows) ----
        const _Float16* Hr = H16 + (size_t)p * (kB * kH);
        const _Float16* Cr = C16 + (size_t)p * (kB * kH);
        #pragma unroll
        for (int it = 0; it < 16; ++it) {
            int idx = it * TPB + tid;             // 0..4095: 64 rows x 64 v8h
            int row = idx >> 6, c8 = idx & 63;
            *(v8h*)(sXH + row * kXHrow + kI + c8 * 8) =
                *(const v8h*)(Hr + (size_t)row * kH + c8 * 8);
        }
        #pragma unroll
        for (int it = 0; it < 16; ++it) {
            int idx = it * TPB + tid;
            int row = idx >> 6, c8 = idx & 63;
            *(v8h*)(sC + row * kCrow + c8 * 8) =
                *(const v8h*)(Cr + (size_t)row * kH + c8 * 8);
        }
        __syncthreads();

        // ---- opaque (loop-variant) zero offset: blocks LICM from hoisting the
        //      weight-fragment loads across steps while keeping GLOBAL addrspace
        int zofs = 0;
        asm volatile("" : "+v"(zofs));
        const _Float16* wbi = bip0 + zofs;
        const _Float16* wbf = bfp0 + zofs;
        const _Float16* wbg = bgp0 + zofs;
        const _Float16* wbo = bop0 + zofs;
        const _Float16* wbd = bdp0 + zofs;

        v8f ai = {}; v8f af = {}; v8f ag = {}; v8f ao = {}; v8f acs = {};

        // gates: fused K over [x|h] (24 chunks of 32)
        #pragma unroll
        for (int kc = 0; kc < 24; ++kc) {
            v16h a = load_a(sXH + aoff + kc * 32);
            ai = wmma32(a, load_b(wbi + kc * 32), ai);
            af = wmma32(a, load_b(wbf + kc * 32), af);
            ag = wmma32(a, load_b(wbg + kc * 32), ag);
            ao = wmma32(a, load_b(wbo + kc * 32), ao);
        }
        // c_s pre-activation = c @ W_d (16 chunks)
        #pragma unroll
        for (int kc = 0; kc < 16; ++kc) {
            acs = wmma32(load_a(sC + coff + kc * 32), load_b(wbd + kc * 32), acs);
        }

        // ---- local cell update: lane owns 8 batch rows x 1 h column ----
        _Float16* Hw  = H16 + (size_t)(1 - p) * (kB * kH);
        _Float16* Cw  = C16 + (size_t)(1 - p) * (kB * kH);
        const float* C32r = C32 + (size_t)p * (kB * kH);
        float*       C32w = C32 + (size_t)(1 - p) * (kB * kH);

        #pragma unroll
        for (int r = 0; r < 8; ++r) {
            const int b   = rb * 16 + r + up * 8;
            const float tt = tvec[(size_t)tf * kB + b];
            const float cs   = tanhf(acs[r] + bv_d);
            const float cold = C32r[(size_t)b * kH + nlm];
            const float cadj = cold - cs + cs / logf(2.718281828459045f + tt);
            const float ig = sigmoidf_(ai[r] + bv_i);
            const float fg = sigmoidf_(af[r] + bv_f);
            const float og = sigmoidf_(ao[r] + bv_o);
            const float gg = tanhf(ag[r] + bv_g);
            const float cn = fg * cadj + ig * gg;
            const float hn = og * tanhf(cn);
            C32w[(size_t)b * kH + nlm] = cn;
            Cw[(size_t)b * kH + nlm]   = (_Float16)cn;
            Hw[(size_t)b * kH + nlm]   = (_Float16)hn;
            out[((size_t)tf * kB + b) * (2 * kH) + (size_t)dir * kH + nlm] = hn;
        }

        // ---- arrive (state stores + this block's LDS reads are done) ----
        __syncthreads();
        if (tid == 0) {
            __threadfence();             // release h/c/out stores
            atomicAdd(cnt, 1u);
        }
        // overlap next step's x stage with other blocks' progress
        if (s + 1 < kT) copy_x(dir ? (kT - 2 - s) : (s + 1));
    }
}

// ---------------- launch ----------------

extern "C" void kernel_launch(void* const* d_in, const int* in_sizes, int n_in,
                              void* d_out, int out_size, void* d_ws, size_t ws_size,
                              hipStream_t stream) {
    const float* x     = (const float*)d_in[0];
    const float* tvec  = (const float*)d_in[1];
    const float* Wih_f = (const float*)d_in[2];
    const float* Whh_f = (const float*)d_in[3];
    const float* b_f   = (const float*)d_in[4];
    const float* Wd_f  = (const float*)d_in[5];
    const float* bd_f  = (const float*)d_in[6];
    const float* Wih_b = (const float*)d_in[7];
    const float* Whh_b = (const float*)d_in[8];
    const float* b_b   = (const float*)d_in[9];
    const float* Wd_b  = (const float*)d_in[10];
    const float* bd_b  = (const float*)d_in[11];

    char* ws = (char*)d_ws;
    size_t off = 0;
    auto take = [&](size_t bytes) { char* p = ws + off; off += (bytes + 255) & ~size_t(255); return p; };

    _Float16* x16  = (_Float16*)take((size_t)kT * kB * kI * sizeof(_Float16));   // 16 MB
    _Float16* wcat = (_Float16*)take((size_t)2 * kG * kK * sizeof(_Float16));    // 6 MB
    _Float16* wd   = (_Float16*)take((size_t)2 * kH * kH * sizeof(_Float16));    // 1 MB
    size_t st_h16 = (size_t)2 * 2 * kB * kH * sizeof(_Float16);
    size_t st_c16 = st_h16;
    size_t st_c32 = (size_t)2 * 2 * kB * kH * sizeof(float);
    char* stbase = ws + off;
    _Float16* h16 = (_Float16*)take(st_h16);
    _Float16* c16 = (_Float16*)take(st_c16);
    float*    c32 = (float*)take(st_c32);
    unsigned* cnt = (unsigned*)take(256);
    size_t stbytes = (size_t)((ws + off) - stbase);
    int zwords = (int)(stbytes / 4);

    const int nx = kT * kB * kI;
    k_cvt_x<<<(nx + 255) / 256, 256, 0, stream>>>(x, x16, nx);

    const long nw = 2L * kG * kK;
    k_pack_wcat<<<(int)((nw + 255) / 256), 256, 0, stream>>>(Wih_f, Whh_f, Wih_b, Whh_b, wcat);

    const long nd = 2L * kH * kH;
    k_pack_wd<<<(int)((nd + 255) / 256), 256, 0, stream>>>(Wd_f, Wd_b, wd);

    k_zero32<<<(zwords + 255) / 256, 256, 0, stream>>>((unsigned*)stbase, zwords);

    k_lstm<<<NBLK, TPB, kLdsBytes, stream>>>(tvec, b_f, bd_f, b_b, bd_b,
                                             x16, wcat, wd, h16, c16, c32,
                                             (float*)d_out, cnt);
}
